// MultiHeadedAttentionWithBias_54322746360188
// MI455X (gfx1250) — matrix operand order repaired and checked by reading, not compiled, not run on hardware
//
#include <hip/hip_runtime.h>
#include <hip/hip_bf16.h>
#include <math.h>

// Problem constants
#define B_  4
#define S_  1024
#define D_  1024
#define H_  16
#define DK_ 64
#define M_  (B_*S_)   // 4096 tokens
#define K_  D_
#define N_  D_

typedef __attribute__((ext_vector_type(16))) __bf16        v16bf;
typedef __attribute__((ext_vector_type(8)))  float         v8f;
typedef __attribute__((ext_vector_type(4)))  unsigned int  u32x4;
typedef __attribute__((ext_vector_type(4)))  float         f32x4;

union Frag {
    v16bf v;
    u32x4 q[2];
};

__device__ __forceinline__ unsigned short f2bf(float f) {
    union { float f; unsigned int u; } c; c.f = f;
    unsigned int u = c.u;
    unsigned int r = u + 0x7FFFu + ((u >> 16) & 1u);  // round-to-nearest-even
    return (unsigned short)(r >> 16);
}

// reductions confined to each 16-lane half of the wave32
__device__ __forceinline__ float hmax16(float v) {
    v = fmaxf(v, __shfl_xor(v, 1));
    v = fmaxf(v, __shfl_xor(v, 2));
    v = fmaxf(v, __shfl_xor(v, 4));
    v = fmaxf(v, __shfl_xor(v, 8));
    return v;
}
__device__ __forceinline__ float hsum16(float v) {
    v += __shfl_xor(v, 1);
    v += __shfl_xor(v, 2);
    v += __shfl_xor(v, 4);
    v += __shfl_xor(v, 8);
    return v;
}

// CDNA5 async global->LDS copy (ASYNCcnt-tracked); INST_OFFSET applies to both
// the LDS and global addresses, so one address pair serves 4 chunks.
__device__ __forceinline__ void async_g2l_4x16(unsigned lds_addr, const void* gaddr) {
    asm volatile("global_load_async_to_lds_b128 %0, %1, off"
                 :: "v"(lds_addr), "v"(gaddr) : "memory");
    asm volatile("global_load_async_to_lds_b128 %0, %1, off offset:16"
                 :: "v"(lds_addr), "v"(gaddr) : "memory");
    asm volatile("global_load_async_to_lds_b128 %0, %1, off offset:32"
                 :: "v"(lds_addr), "v"(gaddr) : "memory");
    asm volatile("global_load_async_to_lds_b128 %0, %1, off offset:48"
                 :: "v"(lds_addr), "v"(gaddr) : "memory");
}
__device__ __forceinline__ void wait_async0() {
    asm volatile("s_wait_asynccnt 0" ::: "memory");
}

// ---------------------------------------------------------------- f32 -> bf16
__global__ void cvt_bf16_k(const float* __restrict__ src,
                           unsigned short* __restrict__ dst, int n) {
    int i = blockIdx.x * blockDim.x + threadIdx.x;
    if (i < n) dst[i] = f2bf(src[i]);
}

// ------------------------------------------------- bf16 WMMA GEMM: A @ W^T + b
// A: [M,K] bf16, W: [N,K] bf16 (both K-major -> identical per-lane load shape)
// Wave tile 64x64 (16 wmma per 16 b128 loads per K-step), block 8 waves 2x4,
// block tile 128x256.
// mode 0: out bf16 [B,H,S,DK]   (Q,K projections)
// mode 1: out bf16 [B,H,DK,S]   (V projection, transposed for P@V B-frags)
// mode 2: out f32  [M,N]        (final output projection)
__global__ __launch_bounds__(256) void gemm_bf16_k(
    const unsigned short* __restrict__ A,
    const unsigned short* __restrict__ W,
    const float* __restrict__ bias,
    unsigned short* __restrict__ out_bf,
    float* __restrict__ out_f,
    int mode)
{
    const int lane = threadIdx.x & 31;
    const int wv   = threadIdx.x >> 5;       // 0..7
    const int ln   = lane & 15;
    const int hi   = lane >> 4;
    const int m_base = blockIdx.y * 128 + (wv >> 2) * 64;
    const int n_base = blockIdx.x * 256 + (wv & 3)  * 64;

    v8f c[4][4] = {};

    for (int kk = 0; kk < K_; kk += 32) {
        Frag a[4], bf[4];
        #pragma unroll
        for (int i = 0; i < 4; ++i) {
            // A-frag: lane row M=ln; K chunks {8*hi..+7} and {16+8*hi..+7}
            const unsigned short* p = A + (size_t)(m_base + i*16 + ln) * K_ + kk + 8*hi;
            a[i].q[0] = *(const u32x4*)p;
            a[i].q[1] = *(const u32x4*)(p + 16);
        }
        #pragma unroll
        for (int j = 0; j < 4; ++j) {
            // B-frag: lane col N=ln; K chunks {8*hi..+7} and {16+8*hi..+7}
            const unsigned short* p = W + (size_t)(n_base + j*16 + ln) * K_ + kk + 8*hi;
            bf[j].q[0] = *(const u32x4*)p;
            bf[j].q[1] = *(const u32x4*)(p + 16);
        }
        #pragma unroll
        for (int i = 0; i < 4; ++i)
            #pragma unroll
            for (int j = 0; j < 4; ++j)
                c[i][j] = __builtin_amdgcn_wmma_f32_16x16x32_bf16(
                    false, a[i].v, false, bf[j].v, (short)0, c[i][j], false, false);
    }

    // epilogue: C-frag element (m = r + 8*hi, n = ln)
    #pragma unroll
    for (int i = 0; i < 4; ++i) {
        #pragma unroll
        for (int j = 0; j < 4; ++j) {
            #pragma unroll
            for (int r = 0; r < 8; ++r) {
                int m = m_base + i*16 + r + 8*hi;
                int n = n_base + j*16 + ln;
                float val = c[i][j][r] + bias[n];
                int b = m >> 10, s = m & (S_-1);
                int h = n >> 6,  d = n & (DK_-1);
                if (mode == 0) {
                    out_bf[((size_t)(b*H_ + h)*S_ + s)*DK_ + d] = f2bf(val);
                } else if (mode == 1) {
                    out_bf[((size_t)(b*H_ + h)*DK_ + d)*S_ + s] = f2bf(val);
                } else {
                    out_f[(size_t)m*N_ + n] = val;
                }
            }
        }
    }
}

// --------------------------------------------------------------- attention
// One block = 16 waves = all 16 heads of one (b, 16-query tile), so the
// [B,S,S,H] bias tile (h contiguous) is staged into LDS with coalesced,
// ASYNC global->LDS b128 copies (overlapped with the Q@K^T WMMAs) and every
// byte of each cacheline is consumed.
// Qp,Kp: [B,H,S,DK] bf16.  Vt: [B,H,DK,S] bf16.  ctx out: [B,S,D] bf16.
#define BKSTR 20   // dwords per (q,k) cell: 16 h values + 4 pad (16B aligned,
                   // 20*ln mod 64 conflict-free over a 16-lane half)
__global__ __launch_bounds__(512) void attn_k(
    const unsigned short* __restrict__ Qp,
    const unsigned short* __restrict__ Kp,
    const unsigned short* __restrict__ Vt,
    const float* __restrict__ bias,
    const unsigned char* __restrict__ mask,
    unsigned short* __restrict__ ctx)
{
    __shared__ float          Bt[16 * 32 * BKSTR];   // 40 KB bias tile
    __shared__ unsigned short Pl[16][16 * 32];       // 16 KB per-wave P tiles

    const int tid  = threadIdx.x;
    const int lane = tid & 31;
    const int wv   = tid >> 5;        // 0..15 == head
    const int ln   = lane & 15;
    const int hi   = lane >> 4;
    const int b    = blockIdx.x;
    const int q0   = blockIdx.y * 16;
    const int h    = wv;
    const int bh   = b * H_ + h;

    const unsigned short* Qb = Qp + ((size_t)bh * S_ + q0) * DK_;
    const unsigned short* Kb = Kp + (size_t)bh * S_ * DK_;
    const unsigned short* Vb = Vt + (size_t)bh * DK_ * S_;

    // Q tile 16x64 -> two A-frags, held for the whole row strip
    Frag qf0, qf1;
    {
        const unsigned short* p = Qb + (size_t)ln * DK_;
        qf0.q[0] = *(const u32x4*)(p +      8*hi);
        qf0.q[1] = *(const u32x4*)(p + 16 + 8*hi);
        qf1.q[0] = *(const u32x4*)(p + 32 + 8*hi);
        qf1.q[1] = *(const u32x4*)(p + 48 + 8*hi);
    }

    bool mq[8];
    #pragma unroll
    for (int r = 0; r < 8; ++r) mq[r] = mask[b*S_ + q0 + r + 8*hi] != 0;

    float m_s[8], l_s[8];
    #pragma unroll
    for (int r = 0; r < 8; ++r) { m_s[r] = -INFINITY; l_s[r] = 0.f; }
    v8f o[4] = {};

    // staging map: thread -> (q row, k column) of the bias tile; each thread
    // async-copies its 64B cell (16 h values) straight into LDS.
    const int sq = tid >> 5;          // 0..15
    const int sk = tid & 31;          // 0..31
    const float*   bias_row = bias + ((size_t)(b*S_ + q0 + sq) * S_ + sk) * H_;
    const unsigned Bdst     = (unsigned)(size_t)(Bt + (sq*32 + sk) * BKSTR);

    const float C2 = 0.18033688011112042f;   // (1/sqrt(64)) * log2(e)

    for (int kb = 0; kb < S_; kb += 32) {
        // ---- async bias staging: 32 KB block-wide, engine runs during WMMAs
        async_g2l_4x16(Bdst, bias_row + (size_t)kb * H_);
        if (kb + 32 < S_)
            __builtin_prefetch(Kb + (size_t)(kb + 32 + ln)*DK_, 0, 1);

        // ---- scores (independent of bias tile): 16x32 of Q @ K^T
        v8f acc0 = {}, acc1 = {};
        {
            const unsigned short* p = Kb + (size_t)(kb + ln) * DK_;
            Frag k00, k01;
            k00.q[0] = *(const u32x4*)(p + 8*hi);
            k00.q[1] = *(const u32x4*)(p + 16 + 8*hi);
            k01.q[0] = *(const u32x4*)(p + 32 + 8*hi);
            k01.q[1] = *(const u32x4*)(p + 48 + 8*hi);
            acc0 = __builtin_amdgcn_wmma_f32_16x16x32_bf16(false, qf0.v, false, k00.v, (short)0, acc0, false, false);
            acc0 = __builtin_amdgcn_wmma_f32_16x16x32_bf16(false, qf1.v, false, k01.v, (short)0, acc0, false, false);
        }
        {
            const unsigned short* p = Kb + (size_t)(kb + 16 + ln) * DK_;
            Frag k10, k11;
            k10.q[0] = *(const u32x4*)(p + 8*hi);
            k10.q[1] = *(const u32x4*)(p + 16 + 8*hi);
            k11.q[0] = *(const u32x4*)(p + 32 + 8*hi);
            k11.q[1] = *(const u32x4*)(p + 48 + 8*hi);
            acc1 = __builtin_amdgcn_wmma_f32_16x16x32_bf16(false, qf0.v, false, k10.v, (short)0, acc1, false, false);
            acc1 = __builtin_amdgcn_wmma_f32_16x16x32_bf16(false, qf1.v, false, k11.v, (short)0, acc1, false, false);
        }

        wait_async0();     // own async copies have landed in LDS
        __syncthreads();   // whole bias tile visible to all heads

        // ---- bias + mask + online softmax (base-2 domain)
        const bool mk0 = mask[b*S_ + kb + ln]      != 0;
        const bool mk1 = mask[b*S_ + kb + 16 + ln] != 0;
        float scale_r[8];
        #pragma unroll
        for (int r = 0; r < 8; ++r) {
            int qr = r + 8*hi;
            float b0 = Bt[(qr*32 + ln)*BKSTR + h];
            float b1 = Bt[(qr*32 + 16 + ln)*BKSTR + h];
            float x0 = (mq[r] && mk0) ? (acc0[r] + b0) * C2 : -1e9f;
            float x1 = (mq[r] && mk1) ? (acc1[r] + b1) * C2 : -1e9f;
            float mx   = hmax16(fmaxf(x0, x1));
            float mnew = fmaxf(m_s[r], mx);
            float sc   = exp2f(m_s[r] - mnew);
            float p0   = exp2f(x0 - mnew);
            float p1   = exp2f(x1 - mnew);
            l_s[r] = l_s[r] * sc + hsum16(p0 + p1);
            m_s[r] = mnew;
            scale_r[r] = sc;
            Pl[wv][qr*32 + ln]      = f2bf(p0);
            Pl[wv][qr*32 + 16 + ln] = f2bf(p1);
        }
        #pragma unroll
        for (int t = 0; t < 4; ++t)
            #pragma unroll
            for (int r = 0; r < 8; ++r)
                o[t][r] *= scale_r[r];

        // ---- P @ V : P A-frag via per-wave LDS round trip
        Frag pf;
        pf.q[0] = *(const u32x4*)&Pl[wv][ln*32 + 8*hi];
        pf.q[1] = *(const u32x4*)&Pl[wv][ln*32 + 16 + 8*hi];
        #pragma unroll
        for (int t = 0; t < 4; ++t) {
            Frag vf;
            const unsigned short* p = Vb + (size_t)(t*16 + ln)*S_ + kb + 8*hi;
            vf.q[0] = *(const u32x4*)p;
            vf.q[1] = *(const u32x4*)(p + 16);
            o[t] = __builtin_amdgcn_wmma_f32_16x16x32_bf16(false, pf.v, false, vf.v, (short)0, o[t], false, false);
        }

        __syncthreads();   // everyone done with Bt before next async overwrite
    }

    // ---- normalize and write context, heads concatenated -> [B,S,D]
    #pragma unroll
    for (int t = 0; t < 4; ++t) {
        #pragma unroll
        for (int r = 0; r < 8; ++r) {
            int s = q0 + r + 8*hi;
            float val = o[t][r] / l_s[r];
            ctx[(size_t)(b*S_ + s)*D_ + h*DK_ + t*16 + ln] = f2bf(val);
        }
    }
}

// ---------------------------------------------------------------------- launch
extern "C" void kernel_launch(void* const* d_in, const int* in_sizes, int n_in,
                              void* d_out, int out_size, void* d_ws, size_t ws_size,
                              hipStream_t stream) {
    (void)in_sizes; (void)n_in; (void)out_size; (void)ws_size;

    const float*         q_in = (const float*)d_in[0];
    const float*         k_in = (const float*)d_in[1];
    const float*         v_in = (const float*)d_in[2];
    const float*         bias = (const float*)d_in[3];
    const unsigned char* mask = (const unsigned char*)d_in[4];
    const float* Wq = (const float*)d_in[5];
    const float* bq = (const float*)d_in[6];
    const float* Wk = (const float*)d_in[7];
    const float* bk = (const float*)d_in[8];
    const float* Wv = (const float*)d_in[9];
    const float* bv = (const float*)d_in[10];
    const float* Wo = (const float*)d_in[11];
    const float* bo = (const float*)d_in[12];

    const size_t NE = (size_t)M_ * D_;   // 4,194,304 activation elements
    const size_t WE = (size_t)D_ * D_;   // 1,048,576 weight elements
    unsigned short* ws  = (unsigned short*)d_ws;
    unsigned short* xq  = ws;
    unsigned short* xk  = xq  + NE;
    unsigned short* xv  = xk  + NE;
    unsigned short* wqb = xv  + NE;
    unsigned short* wkb = wqb + WE;
    unsigned short* wvb = wkb + WE;
    unsigned short* wob = wvb + WE;
    unsigned short* qp  = wob + WE;      // [B,H,S,DK]
    unsigned short* kp  = qp  + NE;      // [B,H,S,DK]
    unsigned short* vt  = kp  + NE;      // [B,H,DK,S]
    unsigned short* ctx = vt  + NE;      // [B,S,D]

    // f32 -> bf16 staging
    const int thr = 256;
    cvt_bf16_k<<<(int)((NE + thr - 1)/thr), thr, 0, stream>>>(q_in, xq, (int)NE);
    cvt_bf16_k<<<(int)((NE + thr - 1)/thr), thr, 0, stream>>>(k_in, xk, (int)NE);
    cvt_bf16_k<<<(int)((NE + thr - 1)/thr), thr, 0, stream>>>(v_in, xv, (int)NE);
    cvt_bf16_k<<<(int)((WE + thr - 1)/thr), thr, 0, stream>>>(Wq, wqb, (int)WE);
    cvt_bf16_k<<<(int)((WE + thr - 1)/thr), thr, 0, stream>>>(Wk, wkb, (int)WE);
    cvt_bf16_k<<<(int)((WE + thr - 1)/thr), thr, 0, stream>>>(Wv, wvb, (int)WE);
    cvt_bf16_k<<<(int)((WE + thr - 1)/thr), thr, 0, stream>>>(Wo, wob, (int)WE);

    dim3 ggrid(N_/256, M_/128);          // (4, 32)
    gemm_bf16_k<<<ggrid, 256, 0, stream>>>(xq, wqb, bq, qp, nullptr, 0);
    gemm_bf16_k<<<ggrid, 256, 0, stream>>>(xk, wkb, bk, kp, nullptr, 0);
    gemm_bf16_k<<<ggrid, 256, 0, stream>>>(xv, wvb, bv, vt, nullptr, 1);

    attn_k<<<dim3(B_, S_/16), 512, 0, stream>>>(qp, kp, vt, bias, mask, ctx);

    gemm_bf16_k<<<ggrid, 256, 0, stream>>>(ctx, wob, bo, nullptr, (float*)d_out, 2);
}
